// EnhancedSelfAttention_76287209112195
// MI455X (gfx1250) — matrix operand order repaired and checked by reading, not compiled, not run on hardware
//
#include <hip/hip_runtime.h>

// ---------------------------------------------------------------------------
// Self-attention for MI455X (gfx1250, wave32, WMMA, TDM).
//   q = x Wq^T + bq ; k = x Wk^T + bk ; v = x Wv^T + bv
//   out = softmax(q k^T / 32) v
// All matmuls via v_wmma_f32_16x16x32_bf16 (f32 accumulate). V is written
// transposed (vT[b][e][s]) so every WMMA B-fragment in the whole pipeline is a
// contiguous 32-byte load. B-fragments are batch-loaded into distinct
// registers ahead of each WMMA chain so loads overlap the matrix pipe
// (partial s_wait_loadcnt instead of drain-to-zero).
// ---------------------------------------------------------------------------

typedef __attribute__((ext_vector_type(16))) __bf16        v16bf;
typedef __attribute__((ext_vector_type(8)))  float         v8f;
typedef __attribute__((ext_vector_type(4)))  unsigned int  v4u;
typedef __attribute__((ext_vector_type(8)))  int           v8i;
typedef __attribute__((ext_vector_type(4)))  int           v4i;

#define EMB   1024
#define SEQ   2048
#define BATCH 4
#define NROWS (BATCH * SEQ)      // 8192 flattened (b,s) rows
#define SCALE 0.03125f           // 1/sqrt(1024)

#if defined(__has_builtin)
#if __has_builtin(__builtin_amdgcn_tensor_load_to_lds)
#define HAVE_TDM 1
#endif
#endif

// ---- WMMA fragment loaders (contiguous k = 16*(lane/16) + s in A and B alike) ------------------

// A tile: 16(M) x 32(K), memory row-major with leading dim `ld` (elements).
__device__ __forceinline__ v16bf load_a_frag(const __bf16* base, int ld) {
  int lane = threadIdx.x & 31;
  int m  = lane & 15;
  int kb = (lane >> 4) << 4;           // 0 or 16
  const __bf16* row = base + m * ld;
  v16bf a;
#pragma unroll
  for (int i = 0; i < 8; ++i) a[i]     = row[kb + i];
#pragma unroll
  for (int i = 0; i < 8; ++i) a[8 + i] = row[kb + 8 + i];
  return a;
}

// B tile: 32(K) x 16(N) where memory holds B^T row-major (N rows of K
// contiguous elements). Matches W (y = x W^T), K (q k^T), and vT (P v).
__device__ __forceinline__ v16bf load_b_frag_rowk(const __bf16* base, int ld) {
  int lane = threadIdx.x & 31;
  int n  = lane & 15;
  int kb = (lane >> 4) << 4;           // 0 or 16
  const __bf16* row = base + n * ld + kb;
  v16bf b;
#pragma unroll
  for (int i = 0; i < 16; ++i) b[i] = row[i];
  return b;
}

// ---- Kernel 1: fp32 -> bf16 ------------------------------------------------
__global__ void f32_to_bf16_kernel(const float* __restrict__ in,
                                   __bf16* __restrict__ out, int n) {
  int i = blockIdx.x * blockDim.x + threadIdx.x;
  int stride = gridDim.x * blockDim.x;
  for (; i < n; i += stride) out[i] = (__bf16)in[i];
}

// ---- Kernel 2: QKV projection ---------------------------------------------
// One 16x64 output tile per wave (4 accumulators -> 4x A reuse).
// 3 mats * 512 row-tiles * 16 col-groups = 24576 wave-tasks, 8 waves/block.
__global__ __launch_bounds__(256)
void qkv_proj_kernel(const __bf16* __restrict__ xh,
                     const __bf16* __restrict__ Wqh, const float* __restrict__ bq,
                     const __bf16* __restrict__ Wkh, const float* __restrict__ bk,
                     const __bf16* __restrict__ Wvh, const float* __restrict__ bv,
                     __bf16* __restrict__ qh, __bf16* __restrict__ kh,
                     __bf16* __restrict__ vT) {
  int wid = blockIdx.x * 8 + (threadIdx.x >> 5);
  int mat = wid >> 13;                 // / (512*16)
  int rt  = (wid >> 4) & 511;
  int cg  = wid & 15;

  const __bf16* W    = (mat == 0) ? Wqh : ((mat == 1) ? Wkh : Wvh);
  const float*  bias = (mat == 0) ? bq  : ((mat == 1) ? bk  : bv);

  int lane = threadIdx.x & 31;
  int n  = lane & 15;
  int mo = (lane >> 4) << 3;           // C/D: lanes16-31 hold M = 8..15
  int row0 = rt << 4;
  int col0 = cg << 6;                  // 64 output columns per wave

  v8f acc[4];
#pragma unroll
  for (int t = 0; t < 4; ++t) {
    float bval = bias[col0 + t * 16 + n];
    v8f c = {bval, bval, bval, bval, bval, bval, bval, bval};
    acc[t] = c;
  }

  const __bf16* xrow = xh + (size_t)row0 * EMB;
#pragma unroll 2
  for (int kk = 0; kk < EMB / 32; ++kk) {
    __builtin_prefetch(xrow + (kk + 4) * 32, 0, 3);   // global_prefetch_b8
    // Batch-load all fragments for this K-step, then run the WMMA chain.
    v16bf bb[4];
#pragma unroll
    for (int t = 0; t < 4; ++t)
      bb[t] = load_b_frag_rowk(W + (size_t)(col0 + t * 16) * EMB + kk * 32, EMB);
    v16bf a = load_a_frag(xrow + kk * 32, EMB);
#pragma unroll
    for (int t = 0; t < 4; ++t)
      acc[t] = __builtin_amdgcn_wmma_f32_16x16x32_bf16(false, a, false, bb[t],
                                                       (short)0, acc[t], false, false);
  }

  if (mat < 2) {
    __bf16* outp = (mat == 0) ? qh : kh;
#pragma unroll
    for (int t = 0; t < 4; ++t)
#pragma unroll
      for (int j = 0; j < 8; ++j)
        outp[(size_t)(row0 + mo + j) * EMB + col0 + t * 16 + n] = (__bf16)acc[t][j];
  } else {
    // vT[b][e][s]: the tile's 8 row-consecutive elems are contiguous along s.
    int bb2 = row0 >> 11;
    int s0 = (row0 & 2047) + mo;       // multiple of 8 -> 16B aligned
    __bf16* vrow = vT + (size_t)bb2 * EMB * SEQ + s0;
#pragma unroll
    for (int t = 0; t < 4; ++t) {
      union { __bf16 h[8]; uint4 u; } tmp;
#pragma unroll
      for (int j = 0; j < 8; ++j) tmp.h[j] = (__bf16)acc[t][j];
      *(uint4*)(vrow + (size_t)(col0 + t * 16 + n) * SEQ) = tmp.u;
    }
  }
}

// ---- Kernel 3: fused attention --------------------------------------------
// Block = (batch, 16-query tile). 256 threads = 8 waves.
// LDS: q tile 32KB (TDM-staged) + 16xSEQ f32 scores 128KB + 16xSEQ bf16 probs
// 64KB + reduce scratch  ->  ~225KB of CDNA5's 320KB/WGP.
__global__ __launch_bounds__(256)
void attention_kernel(const __bf16* __restrict__ qh,
                      const __bf16* __restrict__ kh,
                      const __bf16* __restrict__ vT,
                      float* __restrict__ out) {
  extern __shared__ char smem[];
  __bf16* sQ    = (__bf16*)smem;                              // 16 x EMB bf16
  float*  sS    = (float*)(smem + 32768);                     // 16 x SEQ f32
  __bf16* sP    = (__bf16*)(smem + 32768 + 131072);           // 16 x SEQ bf16
  float*  sRed  = (float*)(smem + 32768 + 131072 + 65536);    // 256 f32
  float*  sStat = sRed + 256;                                 // 16 f32

  int b  = blockIdx.x >> 7;            // / 128
  int qt = blockIdx.x & 127;
  int qrow = b * SEQ + qt * 16;
  const __bf16* qbase = qh + (size_t)qrow * EMB;
  const __bf16* kbase = kh + (size_t)b * SEQ * EMB;
  const __bf16* vTb   = vT + (size_t)b * EMB * SEQ;

  int tid  = threadIdx.x;
  int wave = tid >> 5;
  int lane = tid & 31;
  int n  = lane & 15;
  int mo = (lane >> 4) << 3;

  // ---- Stage the q tile (16 x 1024 bf16 = 32KB) into LDS via the TDM. -----
#if HAVE_TDM
  if (wave == 0) {
    // Tensor DMA descriptor (CDNA5 ISA 8.3/8.4): 2-D tile, 2-byte elements.
    unsigned long long ga = (unsigned long long)(uintptr_t)qbase;
    v4u g0;
    g0[0] = 1u;                                   // count=1 (valid user D#)
    g0[1] = 0u;                                   // lds_addr: sQ at LDS base
    g0[2] = (unsigned)(ga & 0xFFFFFFFFu);         // global_addr[31:0]
    g0[3] = (unsigned)((ga >> 32) & 0x01FFFFFFu)  // global_addr[56:32]
            | (2u << 30);                         // type = 2 ("image")
    v8i g1;
    g1[0] = (1 << 16);          // workgroup_mask=0, data_size=1 (2 bytes)
    g1[1] = (int)(EMB << 16);   // tensor_dim0[15:0] = 1024
    g1[2] = (16 << 16);         // tensor_dim0 hi=0 | tensor_dim1[15:0] = 16
    g1[3] = (int)(EMB << 16);   // tensor_dim1 hi=0 | tile_dim0 = 1024
    g1[4] = 16;                 // tile_dim1 = 16, tile_dim2 = 0
    g1[5] = EMB;                // tensor_dim0_stride = 1024 elems
    g1[6] = 0;
    g1[7] = 0;
    v4i g2 = {0, 0, 0, 0};
    v4i g3 = {0, 0, 0, 0};
#if defined(__clang_major__) && __clang_major__ >= 23
    v8i g4 = {0, 0, 0, 0, 0, 0, 0, 0};
    __builtin_amdgcn_tensor_load_to_lds(g0, g1, g2, g3, g4, 0);
#else
    __builtin_amdgcn_tensor_load_to_lds(g0, g1, g2, g3, 0);
#endif
    __builtin_amdgcn_s_wait_tensorcnt(0);
  }
#else
  {
    const uint4* src = (const uint4*)qbase;
    uint4* dst = (uint4*)sQ;
    for (int i = tid; i < 16 * EMB / 8; i += 256) dst[i] = src[i];
  }
#endif
  __syncthreads();

  // ---- Phase 1: scores = q k^T. Wave w owns key cols [w*256, w*256+256). ---
  for (int pass = 0; pass < 2; ++pass) {
    int cbase = wave * 256 + pass * 128;
    v8f acc[8];
    v8f zero = {0.f, 0.f, 0.f, 0.f, 0.f, 0.f, 0.f, 0.f};
#pragma unroll
    for (int ct = 0; ct < 8; ++ct) acc[ct] = zero;
    for (int kk = 0; kk < EMB / 32; ++kk) {
      // Batch all 8 B-fragments (16 x b128 loads) before the WMMA chain.
      v16bf bb[8];
#pragma unroll
      for (int ct = 0; ct < 8; ++ct)
        bb[ct] = load_b_frag_rowk(kbase + (size_t)(cbase + ct * 16) * EMB + kk * 32, EMB);
      v16bf a = load_a_frag(sQ + kk * 32, EMB);       // from LDS
#pragma unroll
      for (int ct = 0; ct < 8; ++ct)
        acc[ct] = __builtin_amdgcn_wmma_f32_16x16x32_bf16(false, a, false, bb[ct],
                                                          (short)0, acc[ct], false, false);
    }
#pragma unroll
    for (int ct = 0; ct < 8; ++ct)
#pragma unroll
      for (int j = 0; j < 8; ++j)
        sS[(mo + j) * SEQ + cbase + ct * 16 + n] = acc[ct][j];
  }
  __syncthreads();

  // ---- Phase 2: softmax over each of the 16 rows (16 threads per row). ----
  int row = tid >> 4;
  int tpr = tid & 15;
  float m = -3.0e38f;
  for (int c = tpr; c < SEQ; c += 16)
    m = fmaxf(m, sS[row * SEQ + c] * SCALE);
  sRed[row * 16 + tpr] = m;
  __syncthreads();
  if (tpr == 0) {
    float mm = sRed[row * 16];
#pragma unroll
    for (int i = 1; i < 16; ++i) mm = fmaxf(mm, sRed[row * 16 + i]);
    sStat[row] = mm;
  }
  __syncthreads();
  float rmax = sStat[row];
  float lsum = 0.f;
  for (int c = tpr; c < SEQ; c += 16) {
    float e = __expf(sS[row * SEQ + c] * SCALE - rmax);
    sS[row * SEQ + c] = e;
    lsum += e;
  }
  __syncthreads();        // all reads of sStat done before it is overwritten
  sRed[row * 16 + tpr] = lsum;
  __syncthreads();
  if (tpr == 0) {
    float ss = 0.f;
#pragma unroll
    for (int i = 0; i < 16; ++i) ss += sRed[row * 16 + i];
    sStat[row] = 1.0f / ss;
  }
  __syncthreads();
  float rinv = sStat[row];
  for (int c = tpr; c < SEQ; c += 16)
    sP[row * SEQ + c] = (__bf16)(sS[row * SEQ + c] * rinv);
  __syncthreads();

  // ---- Phase 3: out(16 x EMB) = P(16 x SEQ) @ V(SEQ x EMB), V via vT. -----
  int ocol0 = wave * 128;
  v8f zero = {0.f, 0.f, 0.f, 0.f, 0.f, 0.f, 0.f, 0.f};
  v8f oacc[8];
#pragma unroll
  for (int t = 0; t < 8; ++t) oacc[t] = zero;

  for (int kc = 0; kc < SEQ / 32; ++kc) {
    // Batch all 8 B-fragments, then the WMMA chain.
    v16bf bb[8];
#pragma unroll
    for (int t = 0; t < 8; ++t)
      bb[t] = load_b_frag_rowk(vTb + (size_t)(ocol0 + t * 16) * SEQ + kc * 32, SEQ);
    v16bf a = load_a_frag(sP + kc * 32, SEQ);         // from LDS
#pragma unroll
    for (int t = 0; t < 8; ++t)
      oacc[t] = __builtin_amdgcn_wmma_f32_16x16x32_bf16(false, a, false, bb[t],
                                                        (short)0, oacc[t], false, false);
  }

  float* obase = out + (size_t)qrow * EMB;
#pragma unroll
  for (int t = 0; t < 8; ++t)
#pragma unroll
    for (int j = 0; j < 8; ++j)
      obase[(size_t)(mo + j) * EMB + ocol0 + t * 16 + n] = oacc[t][j];
}

// ---------------------------------------------------------------------------
extern "C" void kernel_launch(void* const* d_in, const int* in_sizes, int n_in,
                              void* d_out, int out_size, void* d_ws, size_t ws_size,
                              hipStream_t stream) {
  const float* x  = (const float*)d_in[0];
  const float* Wq = (const float*)d_in[1];
  const float* bq = (const float*)d_in[2];
  const float* Wk = (const float*)d_in[3];
  const float* bk = (const float*)d_in[4];
  const float* Wv = (const float*)d_in[5];
  const float* bv = (const float*)d_in[6];
  float* out = (float*)d_out;

  // Workspace (bf16): xh | Wqh | Wkh | Wvh | qh | kh | vT   (~73.4 MB)
  char* ws = (char*)d_ws;
  __bf16* xh  = (__bf16*)ws; ws += (size_t)NROWS * EMB * 2;
  __bf16* Wqh = (__bf16*)ws; ws += (size_t)EMB * EMB * 2;
  __bf16* Wkh = (__bf16*)ws; ws += (size_t)EMB * EMB * 2;
  __bf16* Wvh = (__bf16*)ws; ws += (size_t)EMB * EMB * 2;
  __bf16* qh  = (__bf16*)ws; ws += (size_t)NROWS * EMB * 2;
  __bf16* kh  = (__bf16*)ws; ws += (size_t)NROWS * EMB * 2;
  __bf16* vT  = (__bf16*)ws; ws += (size_t)NROWS * EMB * 2;  // [b][e][s]

  f32_to_bf16_kernel<<<1024, 256, 0, stream>>>(x,  xh,  NROWS * EMB);
  f32_to_bf16_kernel<<<256,  256, 0, stream>>>(Wq, Wqh, EMB * EMB);
  f32_to_bf16_kernel<<<256,  256, 0, stream>>>(Wk, Wkh, EMB * EMB);
  f32_to_bf16_kernel<<<256,  256, 0, stream>>>(Wv, Wvh, EMB * EMB);

  // 3 mats * 512 row-tiles * 16 col-groups = 24576 waves / 8 per block
  qkv_proj_kernel<<<3072, 256, 0, stream>>>(xh, Wqh, bq, Wkh, bk, Wvh, bv,
                                            qh, kh, vT);

  // 4 batches * 128 query tiles; ~225KB dynamic LDS (CDNA5: 320KB/WGP)
  size_t smem = 32768 + 131072 + 65536 + (256 + 16) * 4;
  attention_kernel<<<BATCH * 128, 256, smem, stream>>>(qh, kh, vT, out);
}
